// PointRCNNBlock_54855322304597
// MI455X (gfx1250) — compile-verified, matrix operand order checked by reading, not run on hardware
//
#include <hip/hip_runtime.h>
#include <math.h>

typedef __bf16 bf16;
typedef __attribute__((ext_vector_type(16))) __bf16 v16bf;
typedef __attribute__((ext_vector_type(8)))  __bf16 v8bf;
typedef __attribute__((ext_vector_type(8)))  float   v8f;

#define BB 4
#define NQ 200
#define CC 128
#define HH 180
#define WW 180
#define NHEAD 8
#define RR 7
#define MROWS (BB*NQ)     // 800

// ---------------------------------------------------------------------------
// CDNA5 async global->LDS copy (ASYNCcnt) + waits
// ---------------------------------------------------------------------------
__device__ __forceinline__ void async_b128(void* lds, const void* gptr) {
  unsigned lds_off = (unsigned)(size_t)lds;   // LDS byte offset (addr[31:0])
  asm volatile("global_load_async_to_lds_b128 %0, %1, off"
               :: "v"(lds_off), "v"(gptr)
               : "memory");
}
__device__ __forceinline__ void wait_async0() {
  asm volatile("s_wait_asynccnt 0x0" ::: "memory");
}
__device__ __forceinline__ void wait_async3() {
  asm volatile("s_wait_asynccnt 0x3" ::: "memory");
}

// ---------------------------------------------------------------------------
// WMMA helpers (CDNA5 v_wmma_f32_16x16x32_bf16, wave32)
// ---------------------------------------------------------------------------
__device__ __forceinline__ v8f wmma_bf16(v16bf a, v16bf b, v8f c) {
  return __builtin_amdgcn_wmma_f32_16x16x32_bf16(false, a, false, b, (short)0, c,
                                                 false, false);
}

// Load one 16x32 bf16 A/B fragment from a row-major [rows][pitch] tile.
// lane l holds row (l&15); kb=(l>>4)*8 -> VGPR0..3 = K[kb..kb+7],
// VGPR4..7 = K[16+kb..16+kb+7] (two contiguous 16B LDS loads).
__device__ __forceinline__ v16bf load_frag(const bf16* base, int row0, int pitch,
                                           int k0, int lane) {
  int r  = row0 + (lane & 15);
  int kb = k0 + ((lane >> 4) << 3);
  const bf16* p = base + (size_t)r * pitch + kb;
  union { v16bf v; v8bf h[2]; } u;
  u.h[0] = *(const v8bf*)(p);
  u.h[1] = *(const v8bf*)(p + 16);
  return u.v;
}

// C/D layout: lane l holds N=(l&15); VGPR r holds M = r + 8*(l>>4).
__device__ __forceinline__ void store_tile_lds(const v8f* acc, float* sF, int mt,
                                               int ntB, int lane) {
  int nloc = lane & 15;
  int m0   = mt * 16 + ((lane >> 4) << 3);
#pragma unroll
  for (int t = 0; t < 4; ++t) {
    int n = (ntB + t) * 16 + nloc;
#pragma unroll
    for (int r = 0; r < 8; ++r) sF[(m0 + r) * 128 + n] = acc[t][r];
  }
}

// ---------------------------------------------------------------------------
// Generic GEMM:  C[M,N] = A[M,K](bf16,row) * W[N,K](bf16,row)^T + bias
// 256 threads (8 waves), tile 64(M) x 128(N), K step 32.
// Double-buffered async global->LDS staging (3 B128 copies / thread / tile).
// EPI 0: fp32 out. EPI 1: bf16 out. EPI 2: scatter dyn params to p1T/p2T.
// Requires: N % 128 == 0, K % 32 == 0. Pad M rows are source-clamped (their
// outputs are masked at the store, so clamping is exact).
// ---------------------------------------------------------------------------
template <int EPI>
__global__ __launch_bounds__(256) void gemm_kernel(
    const bf16* __restrict__ A, const bf16* __restrict__ Bw,
    const float* __restrict__ bias, void* __restrict__ Cout,
    int M, int N, int K) {
  __shared__ alignas(16) bf16 sA[2][64 * 32];
  __shared__ alignas(16) bf16 sB[2][128 * 32];
  const int tid  = threadIdx.x;
  const int lane = tid & 31;
  const int wave = tid >> 5;
  const int m0   = blockIdx.y * 64;
  const int n0   = blockIdx.x * 128;
  const int mt   = wave & 3;
  const int ntB  = (wave >> 2) * 4;

  const int arow = tid >> 2, acol = (tid & 3) << 3;       // A chunk (1/thread)
  const int gm   = min(m0 + arow, M - 1);
  const int brow0 = tid >> 2, bcol0 = (tid & 3) << 3;     // B chunks (2/thread)
  const int brow1 = (tid + 256) >> 2, bcol1 = ((tid + 256) & 3) << 3;

  auto stage = [&](int buf, int k0) {
    async_b128(&sA[buf][arow * 32 + acol], A + (size_t)gm * K + k0 + acol);
    async_b128(&sB[buf][brow0 * 32 + bcol0],
               Bw + (size_t)(n0 + brow0) * K + k0 + bcol0);
    async_b128(&sB[buf][brow1 * 32 + bcol1],
               Bw + (size_t)(n0 + brow1) * K + k0 + bcol1);
  };

  v8f acc[4] = {};
  const int nk = K >> 5;
  stage(0, 0);
  for (int ki = 0; ki < nk; ++ki) {
    const int cur = ki & 1;
    if (ki + 1 < nk) { stage(cur ^ 1, (ki + 1) << 5); wait_async3(); }
    else             { wait_async0(); }
    __syncthreads();
    v16bf af = load_frag(sA[cur], mt * 16, 32, 0, lane);
#pragma unroll
    for (int t = 0; t < 4; ++t) {
      v16bf bfr = load_frag(sB[cur], (ntB + t) * 16, 32, 0, lane);
      acc[t] = wmma_bf16(af, bfr, acc[t]);
    }
    __syncthreads();
  }

  const int nloc = lane & 15;
  const int mr0  = (lane >> 4) << 3;
#pragma unroll
  for (int t = 0; t < 4; ++t) {
    int n = n0 + (ntB + t) * 16 + nloc;
    float bv = bias ? bias[n] : 0.0f;
#pragma unroll
    for (int r = 0; r < 8; ++r) {
      int m = m0 + mt * 16 + mr0 + r;
      if (m >= M) continue;
      float v = acc[t][r] + bv;
      if (EPI == 0) {
        ((float*)Cout)[(size_t)m * N + n] = v;
      } else if (EPI == 1) {
        ((bf16*)Cout)[(size_t)m * N + n] = (bf16)v;
      } else {  // dyn params: store p1^T / p2^T (B-operand layout [n][k])
        int base = n & 16384;
        int rr   = n & 16383;
        int j    = base + ((rr & 127) << 7) + (rr >> 7);
        ((bf16*)Cout)[(size_t)m * 32768 + j] = (bf16)v;
      }
    }
  }
}

// ---------------------------------------------------------------------------
// Small elementwise / prep kernels
// ---------------------------------------------------------------------------
__global__ void f2bf_kernel(const float* __restrict__ in, bf16* __restrict__ out,
                            long n) {
  for (long i = blockIdx.x * (long)blockDim.x + threadIdx.x; i < n;
       i += (long)gridDim.x * blockDim.x)
    out[i] = (bf16)in[i];
}

__global__ void gelu_kernel(const float* __restrict__ in, bf16* __restrict__ out,
                            long n) {
  for (long i = blockIdx.x * (long)blockDim.x + threadIdx.x; i < n;
       i += (long)gridDim.x * blockDim.x) {
    float v = in[i];
    out[i] = (bf16)(0.5f * v * (1.0f + erff(v * 0.70710678118654752f)));
  }
}

// lidar [B,C,H,W] -> featT [B,H,W,C]
__global__ void transpose_kernel(const float* __restrict__ f,
                                 float* __restrict__ ft, long total) {
  for (long i = blockIdx.x * (long)blockDim.x + threadIdx.x; i < total;
       i += (long)gridDim.x * blockDim.x) {
    int c = (int)(i % CC);
    long r = i / CC;
    int x = (int)(r % WW); r /= WW;
    int y = (int)(r % HH);
    int b = (int)(r / HH);
    ft[i] = f[(((size_t)b * CC + c) * HH + y) * WW + x];
  }
}

// query_feat [B,C,NQ] -> x0 [B*NQ, C] (fp32 + bf16)
__global__ __launch_bounds__(128) void xprep_kernel(const float* __restrict__ qf,
                                                    float* __restrict__ x0f,
                                                    bf16* __restrict__ x0b) {
  int n = blockIdx.x, b = blockIdx.y, c = threadIdx.x;
  float v = qf[((size_t)b * CC + c) * NQ + n];
  size_t o = ((size_t)(b * NQ + n)) * CC + c;
  x0f[o] = v;
  x0b[o] = (bf16)v;
}

// ---------------------------------------------------------------------------
// LayerNorm helpers (one row of 128 per 128-thread block)
// ---------------------------------------------------------------------------
__device__ __forceinline__ float block_ln_128(float v, const float* g,
                                              const float* bt, int c,
                                              float* red) {
  red[c] = v; __syncthreads();
  for (int s = 64; s > 0; s >>= 1) { if (c < s) red[c] += red[c + s]; __syncthreads(); }
  float m = red[0] * (1.0f / 128.0f); __syncthreads();
  float vc = v - m;
  red[c] = vc * vc; __syncthreads();
  for (int s = 64; s > 0; s >>= 1) { if (c < s) red[c] += red[c + s]; __syncthreads(); }
  float var = red[0] * (1.0f / 128.0f); __syncthreads();
  return vc * rsqrtf(var + 1e-5f) * g[c] + bt[c];
}

// y = LN(a + b); outputs fp32 and optional bf16
__global__ __launch_bounds__(128) void add_ln_kernel(
    const float* __restrict__ a, const float* __restrict__ bres,
    const float* __restrict__ g, const float* __restrict__ bt,
    float* __restrict__ of, bf16* __restrict__ ob) {
  __shared__ float red[128];
  int r = blockIdx.x, c = threadIdx.x;
  size_t o = (size_t)r * 128 + c;
  float y = block_ln_128(a[o] + bres[o], g, bt, c, red);
  of[o] = y;
  if (ob) ob[o] = (bf16)y;
}

// x2 = LN2( x1 + relu(LN3(t)) )
__global__ __launch_bounds__(128) void dynln_kernel(
    const float* __restrict__ t, const float* __restrict__ x1,
    const float* __restrict__ g3, const float* __restrict__ b3,
    const float* __restrict__ g2, const float* __restrict__ b2,
    float* __restrict__ of, bf16* __restrict__ ob) {
  __shared__ float red[128];
  int r = blockIdx.x, c = threadIdx.x;
  size_t o = (size_t)r * 128 + c;
  float u = block_ln_128(t[o], g3, b3, c, red);
  u = fmaxf(u, 0.0f);
  __syncthreads();
  float y = block_ln_128(x1[o] + u, g2, b2, c, red);
  of[o] = y;
  ob[o] = (bf16)y;
}

// out[b,c,n] = LN3(x2 + ff)
__global__ __launch_bounds__(128) void final_kernel(
    const float* __restrict__ x2, const float* __restrict__ ff,
    const float* __restrict__ g, const float* __restrict__ bt,
    float* __restrict__ out) {
  __shared__ float red[128];
  int r = blockIdx.x, c = threadIdx.x;
  size_t o = (size_t)r * 128 + c;
  float y = block_ln_128(x2[o] + ff[o], g, bt, c, red);
  int b = r / NQ, n = r % NQ;
  out[((size_t)b * CC + c) * NQ + n] = y;
}

// ---------------------------------------------------------------------------
// Attention: one block per (query, head, batch). qkv [B*NQ,384] fp32.
// ---------------------------------------------------------------------------
__global__ __launch_bounds__(64) void attn_kernel(const float* __restrict__ qkv,
                                                  bf16* __restrict__ obf) {
  const int qi = blockIdx.x, h = blockIdx.y, b = blockIdx.z;
  const int t = threadIdx.x;
  __shared__ float qv[16];
  __shared__ float sc[NQ];
  __shared__ float red[64];
  __shared__ float op[64][16];
  const float* base = qkv + ((size_t)(b * NQ + qi)) * 384;
  if (t < 16) qv[t] = base[h * 16 + t] * 0.25f;  // 1/sqrt(16)
  __syncthreads();
  float lmax = -1e30f;
  for (int j = t; j < NQ; j += 64) {
    const float* kr = qkv + ((size_t)(b * NQ + j)) * 384 + 128 + h * 16;
    float d = 0.0f;
#pragma unroll
    for (int e = 0; e < 16; ++e) d += qv[e] * kr[e];
    sc[j] = d;
    lmax = fmaxf(lmax, d);
  }
  red[t] = lmax; __syncthreads();
  for (int s = 32; s > 0; s >>= 1) { if (t < s) red[t] = fmaxf(red[t], red[t + s]); __syncthreads(); }
  float mx = red[0]; __syncthreads();
  float lsum = 0.0f;
  float oacc[16];
#pragma unroll
  for (int e = 0; e < 16; ++e) oacc[e] = 0.0f;
  for (int j = t; j < NQ; j += 64) {
    float e = expf(sc[j] - mx);
    lsum += e;
    const float* vr = qkv + ((size_t)(b * NQ + j)) * 384 + 256 + h * 16;
#pragma unroll
    for (int d = 0; d < 16; ++d) oacc[d] += e * vr[d];
  }
  red[t] = lsum; __syncthreads();
  for (int s = 32; s > 0; s >>= 1) { if (t < s) red[t] += red[t + s]; __syncthreads(); }
  float tot = red[0]; __syncthreads();
#pragma unroll
  for (int d = 0; d < 16; ++d) op[t][d] = oacc[d];
  __syncthreads();
  if (t < 16) {
    float s = 0.0f;
    for (int j = 0; j < 64; ++j) s += op[j][t];
    obf[((size_t)(b * NQ + qi)) * CC + h * 16 + t] = (bf16)(s / tot);
  }
}

// ---------------------------------------------------------------------------
// ROI align (aligned=True, SR=2): featT [B,H,W,C] -> roi bf16 [B*NQ,49,128]
// ---------------------------------------------------------------------------
__global__ __launch_bounds__(128) void roi_kernel(const float* __restrict__ featT,
                                                  const float* __restrict__ boxes,
                                                  bf16* __restrict__ roi) {
  const int q = blockIdx.x, b = blockIdx.y, c = threadIdx.x;
  const float* bx = boxes + ((size_t)(b * NQ + q)) * 4;
  float x0 = bx[0] - 0.5f, y0 = bx[1] - 0.5f;
  float x1 = bx[2] - 0.5f, y1 = bx[3] - 0.5f;
  float bh = (y1 - y0) * (1.0f / RR), bw = (x1 - x0) * (1.0f / RR);
  const float* F = featT + (size_t)b * HH * WW * CC;
  bf16* out = roi + ((size_t)(b * NQ + q)) * 49 * CC;
  for (int iy = 0; iy < RR; ++iy)
    for (int ix = 0; ix < RR; ++ix) {
      float acc = 0.0f;
#pragma unroll
      for (int sy = 0; sy < 2; ++sy)
#pragma unroll
        for (int sx = 0; sx < 2; ++sx) {
          float y = y0 + (iy + (sy + 0.5f) * 0.5f) * bh;
          float x = x0 + (ix + (sx + 0.5f) * 0.5f) * bw;
          bool valid = (y > -1.0f) && (y < (float)HH) && (x > -1.0f) && (x < (float)WW);
          if (!valid) continue;
          y = fminf(fmaxf(y, 0.0f), (float)(HH - 1));
          x = fminf(fmaxf(x, 0.0f), (float)(WW - 1));
          int yi = (int)floorf(y), xi = (int)floorf(x);
          int y2 = min(yi + 1, HH - 1), x2 = min(xi + 1, WW - 1);
          float ly = y - yi, lx = x - xi, hy = 1.0f - ly, hx = 1.0f - lx;
          acc += hy * hx * F[((size_t)yi * WW + xi) * CC + c] +
                 hy * lx * F[((size_t)yi * WW + x2) * CC + c] +
                 ly * hx * F[((size_t)y2 * WW + xi) * CC + c] +
                 ly * lx * F[((size_t)y2 * WW + x2) * CC + c];
        }
      out[(iy * RR + ix) * CC + c] = (bf16)(acc * 0.25f);
    }
}

// ---------------------------------------------------------------------------
// Per-query dynamic conv: f2 = relu(LN2( relu(LN1(roi@p1)) @ p2 ))
// One 256-thread block per query. 48KB LDS, two 64x128x128 WMMA GEMMs,
// all operand staging via async global->LDS copies (ASYNCcnt).
// ---------------------------------------------------------------------------
__global__ __launch_bounds__(256) void dynconv_kernel(
    const bf16* __restrict__ roi, const bf16* __restrict__ pT,
    const float* __restrict__ g1, const float* __restrict__ b1v,
    const float* __restrict__ g2, const float* __restrict__ b2v,
    bf16* __restrict__ f2out) {
  __shared__ alignas(16) unsigned char smem[48 * 1024];
  bf16*  sA = (bf16*)smem;              // 16KB: roi tile, later f1(bf16)
  bf16*  sP = (bf16*)(smem + 16384);    // 32KB: p1^T / p2^T
  float* sF = (float*)(smem + 16384);   // alias: f1/f2 fp32 staging

  const int tid = threadIdx.x, lane = tid & 31, wave = tid >> 5;
  const int qidx = blockIdx.x;
  const bf16* roiQ = roi + (size_t)qidx * 49 * 128;
  const bf16* pQ   = pT + (size_t)qidx * 32768;
  const int mt = wave & 3, ntB = (wave >> 2) * 4;

  // stage roi A (64x128) async; pad rows source-clamped (outputs masked)
#pragma unroll
  for (int i = 0; i < 4; ++i) {
    int idx = tid + i * 256;
    int row = idx >> 4, col = (idx & 15) << 3;
    int srow = min(row, 48);
    async_b128(sA + row * 128 + col, roiQ + srow * 128 + col);
  }
  // stage p1^T (128x128) async
#pragma unroll
  for (int i = 0; i < 8; ++i) {
    int idx = tid + i * 256;
    int row = idx >> 4, col = (idx & 15) << 3;
    async_b128(sP + row * 128 + col, pQ + row * 128 + col);
  }
  wait_async0();
  __syncthreads();

  v8f acc[4] = {};
#pragma unroll
  for (int k0 = 0; k0 < 128; k0 += 32) {
    v16bf af = load_frag(sA, mt * 16, 128, k0, lane);
#pragma unroll
    for (int t = 0; t < 4; ++t) {
      v16bf bfr = load_frag(sP, (ntB + t) * 16, 128, k0, lane);
      acc[t] = wmma_bf16(af, bfr, acc[t]);
    }
  }
  __syncthreads();
  store_tile_lds(acc, sF, mt, ntB, lane);
  __syncthreads();

  // LN1 + relu over last dim -> sA (bf16)
  if (tid < 64) {
    int row = tid;
    float m = 0.0f;
    for (int c = 0; c < 128; ++c) m += sF[row * 128 + c];
    m *= (1.0f / 128.0f);
    float var = 0.0f;
    for (int c = 0; c < 128; ++c) { float d = sF[row * 128 + c] - m; var += d * d; }
    float inv = rsqrtf(var * (1.0f / 128.0f) + 1e-5f);
    for (int c = 0; c < 128; ++c) {
      float u = (sF[row * 128 + c] - m) * inv * g1[c] + b1v[c];
      sA[row * 128 + c] = (bf16)fmaxf(u, 0.0f);
    }
  }
  __syncthreads();

  // stage p2^T async (overwrites sF alias after all LN1 reads done)
#pragma unroll
  for (int i = 0; i < 8; ++i) {
    int idx = tid + i * 256;
    int row = idx >> 4, col = (idx & 15) << 3;
    async_b128(sP + row * 128 + col, pQ + 16384 + row * 128 + col);
  }
  wait_async0();
  __syncthreads();

  v8f acc2[4] = {};
#pragma unroll
  for (int k0 = 0; k0 < 128; k0 += 32) {
    v16bf af = load_frag(sA, mt * 16, 128, k0, lane);
#pragma unroll
    for (int t = 0; t < 4; ++t) {
      v16bf bfr = load_frag(sP, (ntB + t) * 16, 128, k0, lane);
      acc2[t] = wmma_bf16(af, bfr, acc2[t]);
    }
  }
  __syncthreads();
  store_tile_lds(acc2, sF, mt, ntB, lane);
  __syncthreads();

  // LN2 + relu -> global (only valid 49 rows)
  if (tid < 49) {
    int row = tid;
    float m = 0.0f;
    for (int c = 0; c < 128; ++c) m += sF[row * 128 + c];
    m *= (1.0f / 128.0f);
    float var = 0.0f;
    for (int c = 0; c < 128; ++c) { float d = sF[row * 128 + c] - m; var += d * d; }
    float inv = rsqrtf(var * (1.0f / 128.0f) + 1e-5f);
    bf16* dst = f2out + (size_t)qidx * 6272 + row * 128;
    for (int c = 0; c < 128; ++c) {
      float u = (sF[row * 128 + c] - m) * inv * g2[c] + b2v[c];
      dst[c] = (bf16)fmaxf(u, 0.0f);
    }
  }
}

// ---------------------------------------------------------------------------
// Host side
// ---------------------------------------------------------------------------
extern "C" void kernel_launch(void* const* d_in, const int* in_sizes, int n_in,
                              void* d_out, int out_size, void* d_ws, size_t ws_size,
                              hipStream_t stream) {
  const float* query_feat = (const float*)d_in[0];
  const float* lidar_feat = (const float*)d_in[1];
  const float* boxes      = (const float*)d_in[2];
  const float* Wqkv  = (const float*)d_in[3];
  const float* bqkv  = (const float*)d_in[4];
  const float* Wo    = (const float*)d_in[5];
  const float* bo    = (const float*)d_in[6];
  const float* ln1_g = (const float*)d_in[7];
  const float* ln1_b = (const float*)d_in[8];
  const float* ln2_g = (const float*)d_in[9];
  const float* ln2_b = (const float*)d_in[10];
  const float* ln3_g = (const float*)d_in[11];
  const float* ln3_b = (const float*)d_in[12];
  const float* dyn_W = (const float*)d_in[13];
  const float* dyn_b = (const float*)d_in[14];
  const float* lnd1_g = (const float*)d_in[15];
  const float* lnd1_b = (const float*)d_in[16];
  const float* lnd2_g = (const float*)d_in[17];
  const float* lnd2_b = (const float*)d_in[18];
  const float* out_W  = (const float*)d_in[19];
  const float* out_b  = (const float*)d_in[20];
  const float* lnd3_g = (const float*)d_in[21];
  const float* lnd3_b = (const float*)d_in[22];
  const float* W1 = (const float*)d_in[23];
  const float* b1 = (const float*)d_in[24];
  const float* W2 = (const float*)d_in[25];
  const float* b2 = (const float*)d_in[26];

  // workspace carve-out
  char* ws = (char*)d_ws;
  size_t off = 0;
  auto alloc = [&](size_t bytes) -> void* {
    void* p = ws + off;
    off += (bytes + 255) & ~(size_t)255;
    return p;
  };
  float* featT = (float*)alloc((size_t)BB * HH * WW * CC * 4);
  float* x0f   = (float*)alloc((size_t)MROWS * CC * 4);
  bf16*  x0b   = (bf16*)alloc((size_t)MROWS * CC * 2);
  bf16*  Wqkvb = (bf16*)alloc((size_t)384 * 128 * 2);
  bf16*  Wob   = (bf16*)alloc((size_t)128 * 128 * 2);
  bf16*  dynWb = (bf16*)alloc((size_t)32768 * 128 * 2);
  bf16*  outWb = (bf16*)alloc((size_t)128 * 6272 * 2);
  bf16*  W1b   = (bf16*)alloc((size_t)512 * 128 * 2);
  bf16*  W2b   = (bf16*)alloc((size_t)128 * 512 * 2);
  float* qkvF  = (float*)alloc((size_t)MROWS * 384 * 4);
  bf16*  obf   = (bf16*)alloc((size_t)MROWS * CC * 2);
  float* tmpo  = (float*)alloc((size_t)MROWS * CC * 4);
  float* x1f   = (float*)alloc((size_t)MROWS * CC * 4);
  bf16*  x1b   = (bf16*)alloc((size_t)MROWS * CC * 2);
  bf16*  roibf = (bf16*)alloc((size_t)MROWS * 49 * CC * 2);
  bf16*  pT    = (bf16*)alloc((size_t)MROWS * 32768 * 2);
  bf16*  f2buf = (bf16*)alloc((size_t)MROWS * 6272 * 2);
  float* tmpd  = (float*)alloc((size_t)MROWS * CC * 4);
  float* x2f   = (float*)alloc((size_t)MROWS * CC * 4);
  bf16*  x2b   = (bf16*)alloc((size_t)MROWS * CC * 2);
  float* hF    = (float*)alloc((size_t)MROWS * 512 * 4);
  bf16*  hb    = (bf16*)alloc((size_t)MROWS * 512 * 2);
  float* tmpf  = (float*)alloc((size_t)MROWS * CC * 4);

  const int MB = (MROWS + 63) / 64;  // 13 M-tiles of 64

  // prep
  transpose_kernel<<<4096, 256, 0, stream>>>(lidar_feat, featT,
                                             (long)BB * HH * WW * CC);
  xprep_kernel<<<dim3(NQ, BB), 128, 0, stream>>>(query_feat, x0f, x0b);
  f2bf_kernel<<<256, 256, 0, stream>>>(Wqkv, Wqkvb, 384L * 128);
  f2bf_kernel<<<256, 256, 0, stream>>>(Wo, Wob, 128L * 128);
  f2bf_kernel<<<4096, 256, 0, stream>>>(dyn_W, dynWb, 32768L * 128);
  f2bf_kernel<<<1024, 256, 0, stream>>>(out_W, outWb, 128L * 6272);
  f2bf_kernel<<<256, 256, 0, stream>>>(W1, W1b, 512L * 128);
  f2bf_kernel<<<256, 256, 0, stream>>>(W2, W2b, 128L * 512);

  // MHA
  gemm_kernel<0><<<dim3(384 / 128, MB), 256, 0, stream>>>(x0b, Wqkvb, bqkv, qkvF,
                                                          MROWS, 384, 128);
  attn_kernel<<<dim3(NQ, NHEAD, BB), 64, 0, stream>>>(qkvF, obf);
  gemm_kernel<0><<<dim3(1, MB), 256, 0, stream>>>(obf, Wob, bo, tmpo,
                                                  MROWS, 128, 128);
  add_ln_kernel<<<MROWS, 128, 0, stream>>>(x0f, tmpo, ln1_g, ln1_b, x1f, x1b);

  // ROI align
  roi_kernel<<<dim3(NQ, BB), 128, 0, stream>>>(featT, boxes, roibf);

  // dynamic conv
  gemm_kernel<2><<<dim3(32768 / 128, MB), 256, 0, stream>>>(x1b, dynWb, dyn_b, pT,
                                                            MROWS, 32768, 128);
  dynconv_kernel<<<MROWS, 256, 0, stream>>>(roibf, pT, lnd1_g, lnd1_b,
                                            lnd2_g, lnd2_b, f2buf);
  gemm_kernel<0><<<dim3(1, MB), 256, 0, stream>>>(f2buf, outWb, out_b, tmpd,
                                                  MROWS, 128, 6272);
  dynln_kernel<<<MROWS, 128, 0, stream>>>(tmpd, x1f, lnd3_g, lnd3_b, ln2_g, ln2_b,
                                          x2f, x2b);

  // FFN
  gemm_kernel<0><<<dim3(512 / 128, MB), 256, 0, stream>>>(x2b, W1b, b1, hF,
                                                          MROWS, 512, 128);
  gelu_kernel<<<1024, 256, 0, stream>>>(hF, hb, (long)MROWS * 512);
  gemm_kernel<0><<<dim3(1, MB), 256, 0, stream>>>(hb, W2b, b2, tmpf,
                                                  MROWS, 128, 512);
  final_kernel<<<MROWS, 128, 0, stream>>>(x2f, tmpf, ln3_g, ln3_b, (float*)d_out);
}
